// GraphEncoderDecoder_11665131175936
// MI455X (gfx1250) — matrix operand order repaired and checked by reading, not compiled, not run on hardware
//
#include <hip/hip_runtime.h>
#include <hip/hip_bf16.h>

typedef __attribute__((ext_vector_type(16))) _Float16 v16h;
typedef __attribute__((ext_vector_type(8)))  float    v8f;

// ---------------- direct VALID conv (stride 1) + ReLU, kernel size templated ----------------
template <int KS>
__global__ void conv_relu_k(const float* __restrict__ in, const float* __restrict__ w,
                            const float* __restrict__ bias, float* __restrict__ out,
                            int Cin, int Hin, int Win, int Cout,
                            int Hout, int Wout)
{
    int tid = blockIdx.x * blockDim.x + threadIdx.x;
    int total = Cout * Hout * Wout;
    if (tid >= total) return;
    int x = tid % Wout;
    int y = (tid / Wout) % Hout;
    int c = tid / (Wout * Hout);
    float acc = bias[c];
    for (int ci = 0; ci < Cin; ++ci) {
        const float* ip = in + (ci * Hin + y) * Win + x;
        const float* wp = w + (c * Cin + ci) * KS * KS;
        #pragma unroll
        for (int ky = 0; ky < KS; ++ky)
            #pragma unroll
            for (int kx = 0; kx < KS; ++kx)
                acc = fmaf(ip[ky * Win + kx], wp[ky * KS + kx], acc);
    }
    out[tid] = acc > 0.f ? acc : 0.f;
}

// ---------------- transposed conv (stride 1) + ReLU; weight [Cin, Cout, K, K] ----------------
template <int KS>
__global__ void convt_relu_k(const float* __restrict__ in, const float* __restrict__ w,
                             const float* __restrict__ bias, float* __restrict__ out,
                             int Cin, int Hin, int Win, int Cout,
                             int Hout, int Wout)
{
    int tid = blockIdx.x * blockDim.x + threadIdx.x;
    int total = Cout * Hout * Wout;
    if (tid >= total) return;
    int x = tid % Wout;
    int y = (tid / Wout) % Hout;
    int o = tid / (Wout * Hout);
    float acc = bias[o];
    for (int ci = 0; ci < Cin; ++ci) {
        const float* ip = in + ci * Hin * Win;
        const float* wp = w + (ci * Cout + o) * KS * KS;
        #pragma unroll
        for (int ky = 0; ky < KS; ++ky) {
            int p = y - ky;
            if ((unsigned)p >= (unsigned)Hin) continue;
            #pragma unroll
            for (int kx = 0; kx < KS; ++kx) {
                int q = x - kx;
                if ((unsigned)q < (unsigned)Win)
                    acc = fmaf(ip[p * Win + q], wp[ky * KS + kx], acc);
            }
        }
    }
    out[tid] = acc > 0.f ? acc : 0.f;
}

// ---------------- MaxPool2d(4) with first-occurrence argmax ----------------
__global__ void pool4_k(const float* __restrict__ in, float* __restrict__ out,
                        int* __restrict__ idx, int C, int Hin, int Win,
                        int Hout, int Wout)
{
    int tid = blockIdx.x * blockDim.x + threadIdx.x;
    int total = C * Hout * Wout;
    if (tid >= total) return;
    int ox = tid % Wout;
    int oy = (tid / Wout) % Hout;
    int c = tid / (Wout * Hout);
    const float* ip = in + (c * Hin + oy * 4) * Win + ox * 4;
    float best = ip[0];
    int bi = 0;
    #pragma unroll
    for (int k = 1; k < 16; ++k) {
        float v = ip[(k >> 2) * Win + (k & 3)];
        if (v > best) { best = v; bi = k; }   // strict > == first argmax (jnp semantics)
    }
    out[tid] = best;
    idx[tid] = bi;
}

// ---------------- MaxUnpool2d(4) as a gather (no memset needed) ----------------
__global__ void unpool4_k(const float* __restrict__ v, const int* __restrict__ idx,
                          float* __restrict__ out, int C, int Hp, int Wp)
{
    int Ho = Hp * 4, Wo = Wp * 4;
    int tid = blockIdx.x * blockDim.x + threadIdx.x;
    int total = C * Ho * Wo;
    if (tid >= total) return;
    int x = tid % Wo;
    int y = (tid / Wo) % Ho;
    int c = tid / (Wo * Ho);
    int p = (c * Hp + (y >> 2)) * Wp + (x >> 2);
    int k = ((y & 3) << 2) | (x & 3);
    out[tid] = (idx[p] == k) ? v[p] : 0.f;
}

// ---------------- graph attention: per-node s_i, t_i ----------------
__global__ void graph_st_k(const float* __restrict__ g, const float* __restrict__ Ww,
                           const float* __restrict__ Wb, const float* __restrict__ aw,
                           float* __restrict__ s, float* __restrict__ t, int N)
{
    int i = blockIdx.x * blockDim.x + threadIdx.x;
    if (i >= N) return;
    float gi[12];
    #pragma unroll
    for (int d = 0; d < 12; ++d) gi[d] = g[i * 12 + d];
    float si = 0.f, ti = 0.f;
    #pragma unroll
    for (int c = 0; c < 12; ++c) {
        float wg = Wb[c];
        #pragma unroll
        for (int d = 0; d < 12; ++d) wg = fmaf(Ww[c * 12 + d], gi[d], wg);
        si = fmaf(wg, aw[c], si);
        ti = fmaf(wg, aw[12 + c], ti);
    }
    s[i] = si;
    t[i] = ti;
}

// ---------------- r_j = sum_i relu(s_i + t_j + b) (O(N^2), s staged in LDS) ----------------
__global__ void colsum_relu_k(const float* __restrict__ s, const float* __restrict__ t,
                              const float* __restrict__ ab, float* __restrict__ r, int N)
{
    __shared__ float ss[256];
    int j = blockIdx.x * 256 + threadIdx.x;   // grid exactly covers N
    float cj = t[j] + ab[0];
    float acc = 0.f;
    for (int base = 0; base < N; base += 256) {
        __syncthreads();
        ss[threadIdx.x] = s[base + threadIdx.x];
        __syncthreads();
        #pragma unroll 8
        for (int k = 0; k < 256; ++k) {
            float v = ss[k] + cj;
            acc += v > 0.f ? v : 0.f;
        }
    }
    r[j] = acc;
}

// ---------------- u = (1/N) sum_j r_j g_j ; m = u@Ww^T + N*Wb ; latent -> z[2000:2100] ----------------
__global__ void latent_k(const float* __restrict__ g, const float* __restrict__ r,
                         const float* __restrict__ Ww, const float* __restrict__ Wb,
                         const float* __restrict__ Pw, const float* __restrict__ Pb,
                         float* __restrict__ z, int N)
{
    __shared__ float red[256 * 12];
    __shared__ float u[12], m[12];
    int tid = threadIdx.x;
    float acc[12];
    #pragma unroll
    for (int d = 0; d < 12; ++d) acc[d] = 0.f;
    for (int j = tid; j < N; j += 256) {
        float rj = r[j];
        #pragma unroll
        for (int d = 0; d < 12; ++d) acc[d] = fmaf(rj, g[j * 12 + d], acc[d]);
    }
    #pragma unroll
    for (int d = 0; d < 12; ++d) red[tid * 12 + d] = acc[d];
    __syncthreads();
    for (int st = 128; st > 0; st >>= 1) {
        if (tid < st)
            for (int d = 0; d < 12; ++d)
                red[tid * 12 + d] += red[(tid + st) * 12 + d];
        __syncthreads();
    }
    if (tid < 12) u[tid] = red[tid] / (float)N;
    __syncthreads();
    if (tid < 12) {
        float mm = (float)N * Wb[tid];
        #pragma unroll
        for (int d = 0; d < 12; ++d) mm = fmaf(u[d], Ww[tid * 12 + d], mm);
        m[tid] = mm;
    }
    __syncthreads();
    if (tid < 100) {
        float lat = Pb[tid];
        #pragma unroll
        for (int c = 0; c < 12; ++c) lat = fmaf(m[c], Pw[tid * 12 + c], lat);
        z[2000 + tid] = lat;
    }
}

// ---------------- bottleneck GEMV via v_wmma_f32_16x16x32_f16 ----------------
// out[m] = sum_k hw[m,k]*z[k] + hb[m].
// A fragment (16-bit A 16x32, ISA 7.12.2): lane=M(0..15 twice), K in {kbA..kbA+7}u{16+kbA..+7},
// kbA = 8*(lane>=16) -> exactly 4 contiguous float4 loads per lane per chunk.
// B: z staged once in LDS (f16, zero-padded to 2112); column 0 only via lane mask multiply.
__global__ void __launch_bounds__(256)
hidden_wmma_k(const float* __restrict__ hw, const float* __restrict__ hb,
              const float* __restrict__ z, float* __restrict__ out,
              int Kdim, int Mtiles)
{
    __shared__ _Float16 zsh[2112];
    int tid = threadIdx.x;
    for (int k = tid; k < 2112; k += 256)
        zsh[k] = (_Float16)(k < Kdim ? z[k] : 0.f);
    __syncthreads();

    int lane = tid & 31;
    int tile = blockIdx.x * 8 + (tid >> 5);
    if (tile >= Mtiles) return;                  // wave-uniform, after the barrier

    int row  = (tile << 4) + (lane & 15);
    const float* rp = hw + (size_t)row * (size_t)Kdim;
    int hi   = (lane >> 4) & 1;
    int kbA  = hi << 3;                          // 0 / 8
    int kbB  = hi << 4;                          // 0 / 16
    _Float16 hm = (_Float16)(((lane & 15) == 0) ? 1.f : 0.f);

    v8f acc = {};
    int k0 = 0;
    for (; k0 + 32 <= Kdim; k0 += 32) {          // 65 full chunks, zero bounds checks
        float4 f0 = *(const float4*)(rp + k0 + kbA);
        float4 f1 = *(const float4*)(rp + k0 + kbA + 4);
        float4 f2 = *(const float4*)(rp + k0 + kbA + 16);
        float4 f3 = *(const float4*)(rp + k0 + kbA + 20);
        v16h a, b;
        a[0] = (_Float16)f0.x;  a[1] = (_Float16)f0.y;  a[2] = (_Float16)f0.z;  a[3] = (_Float16)f0.w;
        a[4] = (_Float16)f1.x;  a[5] = (_Float16)f1.y;  a[6] = (_Float16)f1.z;  a[7] = (_Float16)f1.w;
        a[8] = (_Float16)f2.x;  a[9] = (_Float16)f2.y;  a[10] = (_Float16)f2.z; a[11] = (_Float16)f2.w;
        a[12] = (_Float16)f3.x; a[13] = (_Float16)f3.y; a[14] = (_Float16)f3.z; a[15] = (_Float16)f3.w;
        #pragma unroll
        for (int e = 0; e < 16; ++e) b[e] = hm * zsh[k0 + ((e & 8) ? 16 : 0) + kbA + (e & 7)];
        acc = __builtin_amdgcn_wmma_f32_16x16x32_f16(
            false, a, false, b, (short)0, acc, false, false);
    }
    if (k0 < Kdim) {                             // tail chunk: clamp addresses (branch-free);
        v16h a, b;                               // k >= Kdim products vanish via zsh zero-pad
        #pragma unroll
        for (int e = 0; e < 16; ++e) {
            int k  = k0 + ((e & 8) ? 16 : 0) + kbA + (e & 7);
            int kc = k < Kdim ? k : Kdim - 1;
            a[e] = (_Float16)rp[kc];
            b[e] = hm * zsh[k + 0 - k + k0 + ((e & 8) ? 16 : 0) + kbA + (e & 7)];
        }
        acc = __builtin_amdgcn_wmma_f32_16x16x32_f16(
            false, a, false, b, (short)0, acc, false, false);
    }
    if ((lane & 15) == 0) {                      // D column 0: lanes 0 (M=0..7), 16 (M=8..15)
        int mb = (tile << 4) + (hi << 3);
        #pragma unroll
        for (int v = 0; v < 8; ++v)
            out[mb + v] = acc[v] + hb[mb + v];
    }
}

extern "C" void kernel_launch(void* const* d_in, const int* in_sizes, int n_in,
                              void* d_out, int out_size, void* d_ws, size_t ws_size,
                              hipStream_t stream)
{
    (void)in_sizes; (void)n_in; (void)out_size; (void)ws_size;
    const float* x   = (const float*)d_in[0];
    const float* g   = (const float*)d_in[1];
    const float* c1w = (const float*)d_in[2];
    const float* c1b = (const float*)d_in[3];
    const float* c2w = (const float*)d_in[4];
    const float* c2b = (const float*)d_in[5];
    const float* c3w = (const float*)d_in[6];
    const float* c3b = (const float*)d_in[7];
    const float* d1w = (const float*)d_in[8];
    const float* d1b = (const float*)d_in[9];
    const float* d2w = (const float*)d_in[10];
    const float* d2b = (const float*)d_in[11];
    const float* d3w = (const float*)d_in[12];
    const float* d3b = (const float*)d_in[13];
    const float* aw  = (const float*)d_in[14];
    const float* ab  = (const float*)d_in[15];
    const float* Ww  = (const float*)d_in[16];
    const float* Wb  = (const float*)d_in[17];
    const float* Pw  = (const float*)d_in[18];
    const float* Pb  = (const float*)d_in[19];
    const float* hw  = (const float*)d_in[20];
    const float* hb  = (const float*)d_in[21];
    float* out = (float*)d_out;

    // workspace layout (floats); encoder buffers reused by symmetric decoder stages
    float* ws = (float*)d_ws;
    size_t o = 0;
    float* y1 = ws + o; o += 2488480;           // conv1 out [10,412,604]  (reused: xu1)
    int*   i1 = (int*)(ws + o); o += 155530;    // pool1 idx [10,103,151]
    float* p1 = ws + o; o += 155530;            // pool1 out               (reused: yd2)
    float* y2 = ws + o; o += 345600;            // conv2 out [25,96,144]   (reused: xu2)
    int*   i2 = (int*)(ws + o); o += 21600;     // pool2 idx [25,24,36]
    float* p2 = ws + o; o += 21600;             // pool2 out               (reused: yd1)
    float* y3 = ws + o; o += 32000;             // conv3 out [50,20,32]    (reused: xu3)
    int*   i3 = (int*)(ws + o); o += 2000;      // pool3 idx [50,5,8]
    float* z  = ws + o; o += 2112;              // bottleneck input [2100] (pool3 out + latent)
    float* sA = ws + o; o += 8192;
    float* tA = ws + o; o += 8192;
    float* rA = ws + o; o += 8192;
    float* z2 = ws + o; o += 2000;              // hidden layer out

    const int B = 256;
    auto gr = [](int n, int b) { return (n + b - 1) / b; };

    // ---- encoder ----
    conv_relu_k<8><<<gr(2488480, B), B, 0, stream>>>(x,  c1w, c1b, y1, 3, 419, 611, 10, 412, 604);
    pool4_k<<<gr(155530, B), B, 0, stream>>>(y1, p1, i1, 10, 412, 604, 103, 151);
    conv_relu_k<8><<<gr(345600, B), B, 0, stream>>>(p1, c2w, c2b, y2, 10, 103, 151, 25, 96, 144);
    pool4_k<<<gr(21600, B), B, 0, stream>>>(y2, p2, i2, 25, 96, 144, 24, 36);
    conv_relu_k<5><<<gr(32000, B), B, 0, stream>>>(p2, c3w, c3b, y3, 25, 24, 36, 50, 20, 32);
    pool4_k<<<gr(2000, B), B, 0, stream>>>(y3, z, i3, 50, 20, 32, 5, 8);   // -> z[0:2000]

    // ---- graph attention, algebraically collapsed to a column-sum ----
    graph_st_k<<<gr(8192, B), B, 0, stream>>>(g, Ww, Wb, aw, sA, tA, 8192);
    colsum_relu_k<<<32, B, 0, stream>>>(sA, tA, ab, rA, 8192);
    latent_k<<<1, B, 0, stream>>>(g, rA, Ww, Wb, Pw, Pb, z, 8192);         // -> z[2000:2100]

    // ---- bottleneck GEMV on the WMMA path (16.8 MB weight stream) ----
    hidden_wmma_k<<<16, 256, 0, stream>>>(hw, hb, z, z2, 2100, 125);

    // ---- decoder (buffers reused) ----
    float* xu3 = y3; float* yd1 = p2; float* xu2 = y2; float* yd2 = p1; float* xu1 = y1;
    unpool4_k<<<gr(32000, B), B, 0, stream>>>(z2, i3, xu3, 50, 5, 8);
    convt_relu_k<5><<<gr(21600, B), B, 0, stream>>>(xu3, d1w, d1b, yd1, 50, 20, 32, 25, 24, 36);
    unpool4_k<<<gr(345600, B), B, 0, stream>>>(yd1, i2, xu2, 25, 24, 36);
    convt_relu_k<8><<<gr(155530, B), B, 0, stream>>>(xu2, d2w, d2b, yd2, 25, 96, 144, 10, 103, 151);
    unpool4_k<<<gr(2488480, B), B, 0, stream>>>(yd2, i1, xu1, 10, 103, 151);
    convt_relu_k<8><<<gr(256009, B), B, 0, stream>>>(xu1, d3w, d3b, out, 10, 412, 604, 1, 419, 611);
}
